// TextClassifierAttentionNetwork_66279935312433
// MI455X (gfx1250) — matrix operand order repaired and checked by reading, not compiled, not run on hardware
//
#include <hip/hip_runtime.h>
#include <hip/hip_bf16.h>

typedef __bf16 bf16;
typedef __attribute__((ext_vector_type(16))) __bf16 v16bf;
typedef __attribute__((ext_vector_type(8)))  __bf16 v8bf;
typedef __attribute__((ext_vector_type(8)))  float  v8f;

#define VOCAB 50000
#define EMB   300
#define EMBP  320       // padded K for WMMA (10 steps of 32)
#define EMBPL 328       // LDS row stride (bank-conflict padding, 16B aligned)
#define HID   128
#define B_    16
#define S_    2048

#define QWAVES 8        // attention: waves (16-query-row tiles) per block
#define KSTRIDE 136     // K tile LDS row stride in bf16 (272B: 16 rows -> 16 bank groups)
#define VSTRIDE 40      // V/P tile LDS row stride in bf16 (80B)
#define PSTRIDE 40

__device__ __forceinline__ v8f wmma_bf16(v16bf a, v16bf b, v8f c) {
  // v_wmma_f32_16x16x32_bf16: D = A(16x32) * B(32x16) + C(16x16 f32)
  return __builtin_amdgcn_wmma_f32_16x16x32_bf16(false, a, false, b, (short)0, c,
                                                 false, false);
}

__device__ __forceinline__ v16bf join16(v8bf lo, v8bf hi) {
  v16bf r;
#pragma unroll
  for (int i = 0; i < 8; i++) { r[i] = lo[i]; r[i + 8] = hi[i]; }
  return r;
}

// ---------------------------------------------------------------------------
// Kernel 0: convert+transpose+pad weights to bf16 [HID x EMBP]; zero ctx accum
// ---------------------------------------------------------------------------
__global__ void prep_kernel(const float* __restrict__ Wk,
                            const float* __restrict__ Wq,
                            const float* __restrict__ Wv,
                            bf16* __restrict__ WqT, bf16* __restrict__ WkT,
                            bf16* __restrict__ WvT, float* __restrict__ ctx) {
  int idx = blockIdx.x * blockDim.x + threadIdx.x;
  int total = 3 * HID * EMBP;
  if (idx < total) {
    int which = idx / (HID * EMBP);
    int rem   = idx % (HID * EMBP);
    int n = rem / EMBP, k = rem % EMBP;
    const float* W = (which == 0) ? Wq : (which == 1) ? Wk : Wv;
    bf16* WT       = (which == 0) ? WqT : (which == 1) ? WkT : WvT;
    float v = (k < EMB) ? W[k * HID + n] : 0.f;   // W is [EMB x HID] row-major
    WT[n * EMBP + k] = (bf16)v;
  }
  if (idx < B_ * HID) ctx[idx] = 0.f;
}

// ---------------------------------------------------------------------------
// Kernel 1: embedding gather -> LDS bf16, QKV projection via WMMA.
// One block = 16 tokens; 8 waves x 3 N-tiles cover 24 (Q|K|V x 8) tiles.
// Q,K stored row-major bf16 [B,S,HID]; V stored transposed bf16 [B,HID,S].
// ---------------------------------------------------------------------------
__global__ void __launch_bounds__(256) qkv_kernel(
    const int* __restrict__ x, const float* __restrict__ emb,
    const bf16* __restrict__ WqT, const bf16* __restrict__ WkT,
    const bf16* __restrict__ WvT,
    const float* __restrict__ bq, const float* __restrict__ bk,
    const float* __restrict__ bvec,
    bf16* __restrict__ Qb, bf16* __restrict__ Kb, bf16* __restrict__ VTb) {
  __shared__ bf16 embLds[16 * EMBPL];
  __shared__ int  tokLds[16];
  int tid = threadIdx.x;
  int rt  = blockIdx.x;
  int b   = rt / (S_ / 16);
  int s0  = (rt % (S_ / 16)) * 16;

  if (tid < 16) tokLds[tid] = x[b * S_ + s0 + tid];
  __syncthreads();
  for (int i = tid; i < 16 * EMB; i += 256) {
    int r = i / EMB, c = i % EMB;
    embLds[r * EMBPL + c] = (bf16)emb[(long)tokLds[r] * EMB + c];
  }
  for (int i = tid; i < 16 * (EMBP - EMB); i += 256) {
    int r = i / (EMBP - EMB), c = EMB + i % (EMBP - EMB);
    embLds[r * EMBPL + c] = (bf16)0.f;
  }
  __syncthreads();

  int wave = tid >> 5, lane = tid & 31;
  int mrow = lane & 15, half = lane >> 4;

#pragma unroll
  for (int q = 0; q < 3; q++) {
    int nt  = wave * 3 + q;
    int sel = nt >> 3;          // 0:Q 1:K 2:V
    int ntl = nt & 7;
    const bf16*  WT   = (sel == 0) ? WqT : (sel == 1) ? WkT : WvT;
    const float* bias = (sel == 0) ? bq  : (sel == 1) ? bk  : bvec;
    int h = ntl * 16 + mrow;    // B-fragment: lane holds column h, halves split K
    v8f acc = {};
#pragma unroll
    for (int ks = 0; ks < EMBP / 32; ks++) {
      const bf16* arow = &embLds[mrow * EMBPL + ks * 32 + half * 8];
      v16bf A  = join16(*(const v8bf*)arow, *(const v8bf*)(arow + 16));
      const bf16* brow = &WT[h * EMBP + ks * 32 + half * 8];
      v16bf Bf = join16(*(const v8bf*)brow, *(const v8bf*)(brow + 16));
      acc = wmma_bf16(A, Bf, acc);
    }
    float bb = bias[h];
#pragma unroll
    for (int r = 0; r < 8; r++) {
      float val = acc[r] + bb;           // D-layout: VGPR r <-> row r + 8*half
      int   srow = s0 + r + half * 8;
      if (sel == 2) {
        VTb[((long)b * HID + h) * S_ + srow] = (bf16)val;   // transposed V
      } else {
        bf16* dst = (sel == 0) ? Qb : Kb;
        dst[((long)b * S_ + srow) * HID + h] = (bf16)val;
      }
    }
  }
}

// ---------------------------------------------------------------------------
// Kernel 2: flash attention. 8 waves/block; each wave owns one 16-query tile.
// K/V tiles (32 keys) staged cooperatively in LDS -> 8x less global traffic.
// Online softmax; context rows reduced straight into ctx[B,HID] via atomics.
// ---------------------------------------------------------------------------
__global__ void __launch_bounds__(QWAVES * 32) attn_kernel(
    const bf16* __restrict__ Qb, const bf16* __restrict__ Kb,
    const bf16* __restrict__ VTb, float* __restrict__ ctx) {
  __shared__ bf16 Klds[32 * KSTRIDE];            // 32 keys x 128 (padded rows)
  __shared__ bf16 Vlds[HID * VSTRIDE];           // 128 h    x 32 keys (padded)
  __shared__ bf16 Plds[QWAVES * 16 * PSTRIDE];   // per-wave P re-layout buffer

  int tid = threadIdx.x, wave = tid >> 5, lane = tid & 31;
  int mrow = lane & 15, half = lane >> 4;
  int b  = blockIdx.y;
  int qt = blockIdx.x * QWAVES + wave;
  bf16* Pmine = &Plds[wave * 16 * PSTRIDE];

  // Q A-fragments: [16 x 128] = 4 k-steps
  const bf16* Qrow = Qb + ((long)b * S_ + qt * 16 + mrow) * HID;
  v16bf aQ[4];
#pragma unroll
  for (int ks = 0; ks < 4; ks++) {
    const bf16* p = Qrow + ks * 32 + half * 8;
    aQ[ks] = join16(*(const v8bf*)p, *(const v8bf*)(p + 16));
  }

  v8f zero = {};
  v8f O[8];
#pragma unroll
  for (int t = 0; t < 8; t++) O[t] = zero;
  float mrun[8], lrun[8];
#pragma unroll
  for (int r = 0; r < 8; r++) { mrun[r] = -1e30f; lrun[r] = 0.f; }

  const float scale = 0.08838834764831845f;   // 1/sqrt(128)
  const bf16* Kbase = Kb  + (long)b * S_ * HID;
  const bf16* Vbase = VTb + (long)b * HID * S_;

  for (int j = 0; j < S_; j += 32) {
    // ---- cooperative stage of K tile [32 x 128] and V tile [128 x 32] ------
#pragma unroll
    for (int rep = 0; rep < 2; rep++) {
      int chunk = tid + rep * 256;             // 512 x 16B chunks (8KB)
      int krow  = chunk >> 4;                  // 16 chunks per 256B K row
      int koff  = (chunk & 15) * 8;
      *(v8bf*)&Klds[krow * KSTRIDE + koff] =
          *(const v8bf*)(Kbase + (long)(j + krow) * HID + koff);
      int hrow  = chunk >> 2;                  // 4 chunks per 64B V row
      int voff  = (chunk & 3) * 8;
      *(v8bf*)&Vlds[hrow * VSTRIDE + voff] =
          *(const v8bf*)(Vbase + (long)hrow * S_ + j + voff);
    }
    if (j + 32 < S_) {                         // prefetch next tile into caches
      __builtin_prefetch(Kbase + (long)(j + 32) * HID + tid * 16, 0, 0);
      __builtin_prefetch(Vbase + (long)(tid >> 1) * S_ + j + 32 + (tid & 1) * 16, 0, 0);
    }
    __syncthreads();

    // ---- scores S[16 x 32] as two 16x16 f32 tiles --------------------------
    v8f S0 = zero, S1 = zero;
#pragma unroll
    for (int ks = 0; ks < 4; ks++) {
      const bf16* kr0 = &Klds[mrow * KSTRIDE + ks * 32 + half * 8];
      v16bf B0 = join16(*(const v8bf*)kr0, *(const v8bf*)(kr0 + 16));
      S0 = wmma_bf16(aQ[ks], B0, S0);
      const bf16* kr1 = &Klds[(16 + mrow) * KSTRIDE + ks * 32 + half * 8];
      v16bf B1 = join16(*(const v8bf*)kr1, *(const v8bf*)(kr1 + 16));
      S1 = wmma_bf16(aQ[ks], B1, S1);
    }

    // ---- online softmax: row r of a tile lives in VGPR r across a half -----
#pragma unroll
    for (int r = 0; r < 8; r++) {
      float s0 = S0[r] * scale, s1 = S1[r] * scale;
      float mx = fmaxf(s0, s1);
#pragma unroll
      for (int off = 1; off < 16; off <<= 1)
        mx = fmaxf(mx, __shfl_xor(mx, off, 32));
      float mnew = fmaxf(mrun[r], mx);
      float corr = __expf(mrun[r] - mnew);
      float p0 = __expf(s0 - mnew), p1 = __expf(s1 - mnew);
      float rs = p0 + p1;
#pragma unroll
      for (int off = 1; off < 16; off <<= 1)
        rs += __shfl_xor(rs, off, 32);
      lrun[r] = lrun[r] * corr + rs;
      mrun[r] = mnew;
      int prow = r + half * 8;                 // D-layout row for this VGPR
      Pmine[prow * PSTRIDE + mrow]      = (bf16)p0;   // re-layout P via LDS
      Pmine[prow * PSTRIDE + 16 + mrow] = (bf16)p1;
#pragma unroll
      for (int t = 0; t < 8; t++) O[t][r] *= corr;
    }

    // ---- P as A-fragment [16 x 32]; O += P @ V -----------------------------
    const bf16* pr = &Pmine[mrow * PSTRIDE + half * 8];
    v16bf AP = join16(*(const v8bf*)pr, *(const v8bf*)(pr + 16));
#pragma unroll
    for (int t = 0; t < 8; t++) {
      const bf16* vr = &Vlds[(t * 16 + mrow) * VSTRIDE + half * 8];
      v16bf BV = join16(*(const v8bf*)vr, *(const v8bf*)(vr + 16));
      O[t] = wmma_bf16(AP, BV, O[t]);
    }
    __syncthreads();   // protect K/V LDS before next stage overwrites
  }

  // finalize: context rows = O/l, then sum over the 16 query rows of this tile
#pragma unroll
  for (int t = 0; t < 8; t++) {
    float colsum = 0.f;
#pragma unroll
    for (int r = 0; r < 8; r++) colsum += O[t][r] / lrun[r];
    colsum += __shfl_xor(colsum, 16, 32);      // other half holds rows 8..15
    if (half == 0) atomicAdd(&ctx[b * HID + t * 16 + mrow], colsum);
  }
}

// ---------------------------------------------------------------------------
// Kernel 3: out[b] = sigmoid(ctx[b,:] . Wfc + bfc)
// ---------------------------------------------------------------------------
__global__ void final_kernel(const float* __restrict__ ctx,
                             const float* __restrict__ Wfc,
                             const float* __restrict__ bfc,
                             float* __restrict__ out) {
  int t = threadIdx.x;
  if (t < B_) {
    float s = bfc[0];
    for (int h = 0; h < HID; h++) s += ctx[t * HID + h] * Wfc[h];
    out[t] = 1.f / (1.f + __expf(-s));
  }
}

extern "C" void kernel_launch(void* const* d_in, const int* in_sizes, int n_in,
                              void* d_out, int out_size, void* d_ws, size_t ws_size,
                              hipStream_t stream) {
  (void)in_sizes; (void)n_in; (void)out_size; (void)ws_size;
  const int*   x    = (const int*)d_in[0];
  /* d_in[1] = length: unused by the reference computation */
  const float* emb  = (const float*)d_in[2];
  const float* Wk   = (const float*)d_in[3];
  const float* bk   = (const float*)d_in[4];
  const float* Wq   = (const float*)d_in[5];
  const float* bq   = (const float*)d_in[6];
  const float* Wv   = (const float*)d_in[7];
  const float* bv   = (const float*)d_in[8];
  const float* Wfc  = (const float*)d_in[9];
  const float* bfc  = (const float*)d_in[10];
  float* out = (float*)d_out;

  char* ws = (char*)d_ws;
  size_t o = 0;
  bf16* Qb  = (bf16*)(ws + o); o += (size_t)B_ * S_ * HID * 2;
  bf16* Kb  = (bf16*)(ws + o); o += (size_t)B_ * S_ * HID * 2;
  bf16* VTb = (bf16*)(ws + o); o += (size_t)B_ * HID * S_ * 2;
  bf16* WqT = (bf16*)(ws + o); o += (size_t)HID * EMBP * 2;
  bf16* WkT = (bf16*)(ws + o); o += (size_t)HID * EMBP * 2;
  bf16* WvT = (bf16*)(ws + o); o += (size_t)HID * EMBP * 2;
  float* ctx = (float*)(ws + o); o += (size_t)B_ * HID * 4;

  int prep_n = 3 * HID * EMBP;
  prep_kernel<<<(prep_n + 255) / 256, 256, 0, stream>>>(Wk, Wq, Wv,
                                                        WqT, WkT, WvT, ctx);
  qkv_kernel<<<B_ * S_ / 16, 256, 0, stream>>>(x, emb, WqT, WkT, WvT,
                                               bq, bk, bv, Qb, Kb, VTb);
  attn_kernel<<<dim3((S_ / 16) / QWAVES, B_), QWAVES * 32, 0, stream>>>(
      Qb, Kb, VTb, ctx);
  final_kernel<<<1, 32, 0, stream>>>(ctx, Wfc, bfc, out);
}
